// Point_Transformer_Last_2826088481365
// MI455X (gfx1250) — matrix operand order repaired and checked by reading, not compiled, not run on hardware
//
#include <hip/hip_runtime.h>
#include <hip/hip_bf16.h>

// ---------------------------------------------------------------------------
// Point Transformer Last — CDNA5 (gfx1250) implementation.
// All GEMMs run on V_WMMA_F32_16X16X32_BF16 (bf16 in, fp32 accumulate).
// Tensors stay fp32 in memory; conversion to bf16 (native V_CVT, RNE)
// happens while staging tiles into LDS. Double-buffered LDS + register-
// staged global prefetch pipelines the k-loop.
// B = C = N = 256 everywhere. Workspace: 5 x 64MB buffers (~320 MB of d_ws).
// ---------------------------------------------------------------------------

typedef __attribute__((ext_vector_type(16))) __bf16 v16bf;
typedef __attribute__((ext_vector_type(8)))  float  v8f;

#define BM 128
#define BN 128
#define BK 32
#define LDK 40   // padded LDS row stride in bf16 elements (80 B, 16B-aligned)

__device__ __forceinline__ unsigned short f2bf(float f) {
  // native fp32 -> bf16 convert (lowers to v_cvt_*bf16_f32, RNE)
  __bf16 h = (__bf16)f;
  return __builtin_bit_cast(unsigned short, h);
}

// Generic batched GEMM: C[b,i,j] = sum_k A(b,i,k) * B(b,k,j)  (+ bias[i])
//  AT:     A stored transposed in memory as [K, M] (row stride M)
//  ABATCH: A has a batch dimension (stride M*K); else shared weights
//  BDIFF:  B operand is (B1 - B2) elementwise (subtraction in registers)
//  BIAS:   add bias[i] (per output row / channel)
// B is always batched [b, K, N] row-major. M, N, K multiples of tiles (256).
template<bool AT, bool ABATCH, bool BDIFF, bool BIAS>
__global__ __launch_bounds__(256) void gemm_bf16_wmma(
    const float* __restrict__ Ag, const float* __restrict__ Bg,
    const float* __restrict__ B2g, const float* __restrict__ bias,
    float* __restrict__ Cg, int M, int N, int K)
{
  __shared__ __align__(16) unsigned short As[2][BM * LDK];  // [m][k]
  __shared__ __align__(16) unsigned short Bs[2][BN * LDK];  // [n][k]

  const int b  = blockIdx.z;
  const int m0 = blockIdx.y * BM;
  const int n0 = blockIdx.x * BN;

  const float* Ab  = Ag + (ABATCH ? (size_t)b * M * K : (size_t)0);
  const float* Bb  = Bg + (size_t)b * K * N;
  const float* B2b = BDIFF ? (B2g + (size_t)b * K * N) : nullptr;
  float*       Cb  = Cg + (size_t)b * M * N;

  const int t    = threadIdx.x;
  const int lane = t & 31;
  const int w    = t >> 5;    // 8 waves
  const int wo   = w & 1;     // 2 waves along M (64 rows each)
  const int wn   = w >> 1;    // 4 waves along N (32 cols each)
  const int half = lane >> 4; // 0 / 1 (lanes 0-15 vs 16-31)
  const int r    = lane & 15;

  // ---- loader addressing: each thread fetches 16 contiguous floats for A
  //      and 16 for B per k-step; only the LDS scatter pattern differs. ----
  const int a_row = t >> 1;           // !AT: 0..127
  const int a_ks  = (t & 1) * 16;     // !AT: 0 or 16
  const int a_kk  = t >> 3;           //  AT: 0..31
  const int a_is  = (t & 7) * 16;     //  AT: 0..112
  const float* aptr;
  size_t a_step;
  if (!AT) { aptr = Ab + (size_t)(m0 + a_row) * K + a_ks; a_step = BK; }
  else     { aptr = Ab + (size_t)a_kk * M + (m0 + a_is);  a_step = (size_t)BK * M; }

  const int bkk = t >> 3;
  const int bjs = (t & 7) * 16;
  const float* bptr = Bb + (size_t)bkk * N + (n0 + bjs);
  const float* b2p  = BDIFF ? (B2b + (size_t)bkk * N + (n0 + bjs)) : nullptr;
  const size_t b_step = (size_t)BK * N;

  float areg[16], breg[16];

#define LOAD_TILE() do {                                                   \
    _Pragma("unroll") for (int u = 0; u < 16; ++u) areg[u] = aptr[u];      \
    aptr += a_step;                                                        \
    if (BDIFF) {                                                           \
      _Pragma("unroll") for (int u = 0; u < 16; ++u)                       \
        breg[u] = bptr[u] - b2p[u];                                        \
      b2p += b_step;                                                       \
    } else {                                                               \
      _Pragma("unroll") for (int u = 0; u < 16; ++u) breg[u] = bptr[u];    \
    }                                                                      \
    bptr += b_step;                                                        \
  } while (0)

#define STORE_TILE(buf) do {                                               \
    if (!AT) {                                                             \
      _Pragma("unroll") for (int u = 0; u < 16; ++u)                       \
        As[buf][a_row * LDK + a_ks + u] = f2bf(areg[u]);                   \
    } else {                                                               \
      _Pragma("unroll") for (int u = 0; u < 16; ++u)                       \
        As[buf][(a_is + u) * LDK + a_kk] = f2bf(areg[u]);                  \
    }                                                                      \
    _Pragma("unroll") for (int u = 0; u < 16; ++u)                         \
      Bs[buf][(bjs + u) * LDK + bkk] = f2bf(breg[u]);                      \
  } while (0)

  v8f acc[4][2] = {};  // 4 M-subtiles x 2 N-subtiles of 16x16 fp32

  // prologue: stage tile 0
  LOAD_TILE();
  STORE_TILE(0);
  __syncthreads();

  const int ksteps = K / BK;
  int cur = 0;
  for (int ki = 0; ki < ksteps; ++ki) {
    const bool has_next = (ki + 1) < ksteps;
    if (has_next) {
      LOAD_TILE();                       // global loads overlap the WMMAs
      __builtin_prefetch(aptr, 0, 3);    // k+2 slab -> GL2 (speculative)
      __builtin_prefetch(bptr, 0, 3);
    }

    // ---- WMMA burst: 8 x v_wmma_f32_16x16x32_bf16 on buffer `cur` ----
    #pragma unroll
    for (int ni = 0; ni < 2; ++ni) {
      // B fragment: lane holds column n = col, K = half*8 + (0..7), 16 + half*8 + (0..7)
      const int col = wn * 32 + ni * 16 + r;
      v16bf bf;
      const unsigned short* bp = &Bs[cur][col * LDK + half * 8];
      ((uint4*)&bf)[0] = *(const uint4*)(bp);       // K half*8 + 0..7
      ((uint4*)&bf)[1] = *(const uint4*)(bp + 16);  // K 16+half*8+0..7
      #pragma unroll
      for (int mi = 0; mi < 4; ++mi) {
        // A fragment (ISA 16-bit A 16x32 layout):
        //  elem e<8 -> K = half*8 + e;  e>=8 -> K = 16 + half*8 + (e-8)
        const int row = wo * 64 + mi * 16 + r;
        v16bf af;
        const unsigned short* ap = &As[cur][row * LDK + half * 8];
        ((uint4*)&af)[0] = *(const uint4*)(ap);        // K half*8 + 0..7
        ((uint4*)&af)[1] = *(const uint4*)(ap + 16);   // K 16+half*8+0..7
        acc[mi][ni] = __builtin_amdgcn_wmma_f32_16x16x32_bf16(
            false, af, false, bf, (short)0, acc[mi][ni], false, false);
      }
    }

    if (has_next) STORE_TILE(cur ^ 1);   // fill the other buffer
    __syncthreads();
    cur ^= 1;
  }

#undef LOAD_TILE
#undef STORE_TILE

  // ---- store: D VGPR v holds M = sub_m + half*8 + v, N = sub_n + r ----
  #pragma unroll
  for (int mi = 0; mi < 4; ++mi) {
    #pragma unroll
    for (int ni = 0; ni < 2; ++ni) {
      const int col      = n0 + wn * 32 + ni * 16 + r;
      const int row_base = m0 + wo * 64 + mi * 16 + half * 8;
      #pragma unroll
      for (int v = 0; v < 8; ++v) {
        float val = acc[mi][ni][v];
        if (BIAS) val += bias[row_base + v];
        Cb[(size_t)(row_base + v) * N + col] = val;
      }
    }
  }
}

// ---- row softmax over last dim (rows of length 256) ----
__global__ __launch_bounds__(256) void softmax_rows(float* __restrict__ E)
{
  const int Npts = 256;
  const size_t row = blockIdx.x;
  const int t = threadIdx.x;
  float* Er = E + row * Npts;
  const float v = Er[t];
  __shared__ float red[256];
  red[t] = v;
  __syncthreads();
  for (int o = 128; o > 0; o >>= 1) {
    if (t < o) red[t] = fmaxf(red[t], red[t + o]);
    __syncthreads();
  }
  const float mx = red[0];
  __syncthreads();
  const float e = __expf(v - mx);
  red[t] = e;
  __syncthreads();
  for (int o = 128; o > 0; o >>= 1) {
    if (t < o) red[t] += red[t + o];
    __syncthreads();
  }
  Er[t] = e / red[0];
}

// ---- column L1 renorm: attn[:, :, m] /= (1e-9 + sum_n attn[:, n, m]) ----
__global__ __launch_bounds__(256) void col_renorm(float* __restrict__ E)
{
  const int Npts = 256;
  float* Eb = E + (size_t)blockIdx.x * Npts * Npts;
  const int m = threadIdx.x;
  float s = 0.f;
  for (int n = 0; n < Npts; ++n) s += Eb[(size_t)n * Npts + m];
  const float inv = 1.f / (1e-9f + s);
  for (int n = 0; n < Npts; ++n) Eb[(size_t)n * Npts + m] *= inv;
}

// ---- BatchNorm stats over (B, N) per channel: biased var, rstd ----
__global__ __launch_bounds__(256) void bn_stats(
    const float* __restrict__ T, float* __restrict__ mean,
    float* __restrict__ rstd, int Bn, int C, int N)
{
  const int c = blockIdx.x;
  const int t = threadIdx.x;  // = n (N == 256 == blockDim)
  float s = 0.f, ss = 0.f;
  for (int b = 0; b < Bn; ++b) {
    const float v = T[((size_t)b * C + c) * N + t];
    s += v; ss += v * v;
  }
  __shared__ float rs[256], rss[256];
  rs[t] = s; rss[t] = ss;
  __syncthreads();
  for (int o = 128; o > 0; o >>= 1) {
    if (t < o) { rs[t] += rs[t + o]; rss[t] += rss[t + o]; }
    __syncthreads();
  }
  if (t == 0) {
    const float inv = 1.f / ((float)Bn * (float)N);
    const float m = rs[0] * inv;
    const float var = rss[0] * inv - m * m;
    mean[c] = m;
    rstd[c] = rsqrtf(var + 1e-5f);
  }
}

// ---- y = rscale*resid + relu(g*(T-mean)*rstd + beta); optional 2nd sink ----
__global__ __launch_bounds__(256) void bn_apply(
    const float* __restrict__ T, const float* __restrict__ mean,
    const float* __restrict__ rstd, const float* __restrict__ g,
    const float* __restrict__ beta, const float* __restrict__ resid,
    float rscale, float* __restrict__ outp, float* __restrict__ out2,
    size_t out2_bstride, int C, int N)
{
  const int t  = threadIdx.x;
  const int bc = blockIdx.x;
  const int c  = bc % C;
  const size_t i = (size_t)bc * N + t;
  float v = g[c] * (T[i] - mean[c]) * rstd[c] + beta[c];
  v = fmaxf(v, 0.0f);
  if (resid) v += rscale * resid[i];
  outp[i] = v;
  if (out2) {
    const int b = bc / C;
    out2[(size_t)b * out2_bstride + (size_t)c * N + t] = v;
  }
}

// ---- out[b,n,c] = in[b,c,n] (32x32 LDS tiles) ----
__global__ void transpose_bcn(const float* __restrict__ in,
                              float* __restrict__ out, int C, int N)
{
  __shared__ float tile[32][33];
  const int b  = blockIdx.z;
  const int c0 = blockIdx.y * 32;
  const int n0 = blockIdx.x * 32;
  const int tx = threadIdx.x, ty = threadIdx.y;  // 32 x 8
  for (int i = ty; i < 32; i += 8)
    tile[i][tx] = in[((size_t)b * C + c0 + i) * N + n0 + tx];
  __syncthreads();
  for (int i = ty; i < 32; i += 8)
    out[((size_t)b * N + n0 + i) * C + c0 + tx] = tile[tx][i];
}

// ---------------------------------------------------------------------------
// Host orchestration
// ---------------------------------------------------------------------------

static void launch_gemm(int mode, const float* A, const float* B1,
                        const float* B2, const float* bias, float* Cc,
                        hipStream_t s)
{
  const dim3 g(2, 2, 256), b(256);
  const int M = 256, N = 256, K = 256;
  switch (mode) {
    case 0:  // W @ X (weights shared across batch)
      gemm_bf16_wmma<false,false,false,false><<<g,b,0,s>>>(A,B1,nullptr,nullptr,Cc,M,N,K); break;
    case 1:  // W @ X + bias
      gemm_bf16_wmma<false,false,false,true ><<<g,b,0,s>>>(A,B1,nullptr,bias,Cc,M,N,K); break;
    case 2:  // W @ (B1 - B2) + bias (fused subtraction)
      gemm_bf16_wmma<false,false,true ,true ><<<g,b,0,s>>>(A,B1,B2,bias,Cc,M,N,K); break;
    case 3:  // A^T @ B, A batched (energy = Xk^T Xk)
      gemm_bf16_wmma<true ,true ,false,false><<<g,b,0,s>>>(A,B1,nullptr,nullptr,Cc,M,N,K); break;
    case 4:  // A @ B, A batched (Xr = Xv @ attn)
      gemm_bf16_wmma<false,true ,false,false><<<g,b,0,s>>>(A,B1,nullptr,nullptr,Cc,M,N,K); break;
  }
}

extern "C" void kernel_launch(void* const* d_in, const int* in_sizes, int n_in,
                              void* d_out, int out_size, void* d_ws, size_t ws_size,
                              hipStream_t stream)
{
  (void)in_sizes; (void)n_in; (void)out_size; (void)ws_size;
  const int Bb = 256, C = 256, N = 256;

  const float* x_in   = (const float*)d_in[0];
  const float* conv_w = (const float*)d_in[1];   // [6,C,C]
  const float* bn_g   = (const float*)d_in[2];   // [6,C]
  const float* bn_b   = (const float*)d_in[3];   // [6,C]
  const float* wqk    = (const float*)d_in[4];   // [8,C,C]
  const float* wv     = (const float*)d_in[5];   // [8,C,C]
  const float* bv     = (const float*)d_in[6];   // [8,C]
  const float* wt     = (const float*)d_in[7];   // [8,C,C]
  const float* bt     = (const float*)d_in[8];   // [8,C]
  const float* sg     = (const float*)d_in[9];   // [8,C]
  const float* sb     = (const float*)d_in[10];  // [8,C]
  float* out = (float*)d_out;                    // [B, 4C, N]

  const size_t T = (size_t)Bb * C * N;           // 16.7M elems = 64 MB
  float* X  = (float*)d_ws;                      // activation / scratch
  float* X1 = X  + T;
  float* XK = X1 + T;                            // Xk, then Xr
  float* XV = XK + T;                            // Xv, then pre-BN T
  float* E  = XV + T;                            // energy / attn [B,N,N]
  float* mb = E  + T;                            // per-channel mean
  float* rb = mb + C;                            // per-channel rstd

  const dim3 bn_stats_g(C), bn_apply_g(Bb * C), thr(256);

  // offset-attention SA layer: xout = 2*xin + relu(bn(Wt(xin - Xr) + bt))
  auto run_sa = [&](int i, const float* xin, float* xout) {
    launch_gemm(0, wqk + (size_t)i * C * C, xin, nullptr, nullptr, XK, stream);
    launch_gemm(3, XK, XK, nullptr, nullptr, E, stream);                 // energy
    softmax_rows<<<Bb * N, thr, 0, stream>>>(E);
    col_renorm  <<<Bb,     thr, 0, stream>>>(E);
    launch_gemm(1, wv + (size_t)i * C * C, xin, nullptr, bv + (size_t)i * C, XV, stream);
    launch_gemm(4, XV, E, nullptr, nullptr, XK, stream);                 // Xr -> XK
    launch_gemm(2, wt + (size_t)i * C * C, xin, XK, bt + (size_t)i * C, XV, stream);
    bn_stats<<<bn_stats_g, thr, 0, stream>>>(XV, mb, rb, Bb, C, N);
    bn_apply<<<bn_apply_g, thr, 0, stream>>>(XV, mb, rb,
        sg + (size_t)i * C, sb + (size_t)i * C, xin, 2.0f,
        xout, nullptr, 0, C, N);
  };

  // initial convs: x = relu(bn(W @ x))
  launch_gemm(0, conv_w + (size_t)0 * C * C, x_in, nullptr, nullptr, XV, stream);
  bn_stats<<<bn_stats_g, thr, 0, stream>>>(XV, mb, rb, Bb, C, N);
  bn_apply<<<bn_apply_g, thr, 0, stream>>>(XV, mb, rb, bn_g, bn_b,
      nullptr, 0.f, X, nullptr, 0, C, N);

  launch_gemm(0, conv_w + (size_t)1 * C * C, X, nullptr, nullptr, XV, stream);
  bn_stats<<<bn_stats_g, thr, 0, stream>>>(XV, mb, rb, Bb, C, N);
  bn_apply<<<bn_apply_g, thr, 0, stream>>>(XV, mb, rb, bn_g + C, bn_b + C,
      nullptr, 0.f, X1, nullptr, 0, C, N);

  // 4 blocks; current activation lives in X1 at block entry/exit
  for (int blk = 0; blk < 4; ++blk) {
    run_sa(2 * blk, X1, X);                                   // x1 = 2x + bn-path
    transpose_bcn<<<dim3(8, 8, Bb), dim3(32, 8), 0, stream>>>(X, X1, C, N);
    run_sa(2 * blk + 1, X1, X);                               // x11
    launch_gemm(0, conv_w + (size_t)(blk + 2) * C * C, X, nullptr, nullptr, XV, stream);
    bn_stats<<<bn_stats_g, thr, 0, stream>>>(XV, mb, rb, Bb, C, N);
    // write next activation AND the d_out concat slice [:, blk*C:(blk+1)*C, :]
    bn_apply<<<bn_apply_g, thr, 0, stream>>>(XV, mb, rb,
        bn_g + (size_t)(blk + 2) * C, bn_b + (size_t)(blk + 2) * C,
        nullptr, 0.f, X1, out + (size_t)blk * C * N, (size_t)4 * C * N, C, N);
  }
}
